// AttentionSAGE_49323404427605
// MI455X (gfx1250) — matrix operand order repaired and checked by reading, not compiled, not run on hardware
//
#include <hip/hip_runtime.h>
#include <hip/hip_bf16.h>

// ---------------------------------------------------------------------------
// Problem constants (from reference)
// ---------------------------------------------------------------------------
#define NN     8192
#define SS     32
#define FF     256
#define HH     256
#define G4     1024          // 4*H
#define KK     512           // F + H (fused GEMM depth)
#define HEADS  8

// LSTM kernel tiling
#define TM       32          // rows (sequences) per workgroup
#define RT       2           // row tiles of 16 (TM/16)
#define KT       16          // K tiles of 32 (KK/32)
#define ASTRIDE  520         // LDS A-panel stride in halves (512 + 8 pad)

typedef _Float16 v16h __attribute__((ext_vector_type(16)));
typedef _Float16 h8   __attribute__((ext_vector_type(8)));
typedef _Float16 h4   __attribute__((ext_vector_type(4)));
typedef float    v8f  __attribute__((ext_vector_type(8)));

union AFrag { v16h v; h8 h[2]; };

__device__ __forceinline__ float sigf(float x) {
    return 1.0f / (1.0f + __expf(-x));
}

// ---------------------------------------------------------------------------
// Kernel 0: pack [W_ih^T ; W_hh^T] (K=512 x N=1024) into WMMA B-fragment
// layout as f16, and combine biases.  Packed element order is chosen so that
// a lane's 16-half fragment is a contiguous 32-byte chunk:
//   offset = (((ct*16 + kt)*32 + lane)*16 + i) halves
//   value  = Wcat[kt*32 + (lane<16?0:8) + (i<8?i:8+i)][ct*16 + (lane&15)]
// ---------------------------------------------------------------------------
__global__ __launch_bounds__(256) void k_prep(
    const float* __restrict__ Wih_f, const float* __restrict__ Whh_f,
    const float* __restrict__ bih_f, const float* __restrict__ bhh_f,
    const float* __restrict__ Wih_r, const float* __restrict__ Whh_r,
    const float* __restrict__ bih_r, const float* __restrict__ bhh_r,
    _Float16* __restrict__ Wp, float* __restrict__ biasws)
{
    const int dir = blockIdx.y;
    const float* Wih = dir ? Wih_r : Wih_f;
    const float* Whh = dir ? Whh_r : Whh_f;
    const float* bih = dir ? bih_r : bih_f;
    const float* bhh = dir ? bhh_r : bhh_f;

    const int tid  = blockIdx.x * 256 + threadIdx.x;   // 0 .. 524287
    const int i    = tid & 15;                         // half index in fragment
    const int l    = (tid >> 4) & 31;                  // lane
    const int frag = tid >> 9;                         // ct*16 + kt
    const int kt   = frag & 15;
    const int ct   = frag >> 4;

    const int k   = kt * 32 + ((l & 16) ? 8 : 0) + (i < 8 ? i : 8 + i); // B layout: table order (8h+i | 16+8h+i-8)
    const int col = ct * 16 + (l & 15);                // output gate column

    const float v = (k < FF) ? Wih[col * FF + k] : Whh[col * HH + (k - FF)];
    Wp[(size_t)dir * (KK * G4) + tid] = (_Float16)v;

    if (tid < G4) biasws[dir * G4 + tid] = bih[tid] + bhh[tid];
}

// ---------------------------------------------------------------------------
// Kernel 1: fused bidirectional LSTM recurrence with WMMA gate GEMMs.
// grid = (NN/TM, 2 directions), block = 256 threads = 8 waves.
// Wave w owns gate column-tiles ct = w + 8*m  (m = 0..7)  ==>  for hidden dim
// j = w*16 + c0 + 128*mm (mm = 0,1), gates i/f/g/o sit in accumulators
// m = mm, mm+2, mm+4, mm+6 of the SAME lane -> register-only cell update.
//
// Addressing is arranged so the hot loop is saddr/base + 32-bit lane offset +
// compile-time immediates only (no per-fragment 64-bit VALU math).
// ---------------------------------------------------------------------------
__global__ __launch_bounds__(256) void k_lstm(
    const float* __restrict__ nx,          // [N,S,F] f32
    const _Float16* __restrict__ Wpack,    // packed weights, 2 dirs
    const float* __restrict__ biasws,      // [2][1024]
    _Float16* __restrict__ outr)           // [N,S,512] f16 (fwd|bwd)
{
    const int dir  = blockIdx.y;
    const int n0   = blockIdx.x * TM;
    const int t    = threadIdx.x;
    const int wave = t >> 5;
    const int lane = t & 31;
    const int c0   = lane & 15;
    const int kb_a = (lane & 16) ? 8 : 0;   // A-fragment K sub-base
    const int rowb = (lane & 16) ? 8 : 0;   // C-fragment row base

    __shared__ _Float16 Asm[TM * ASTRIDE];  // [x_s | h] panel, f16

    // Uniform bases + per-lane 32-bit offsets -> saddr-form loads/stores with
    // immediate offsets in the unrolled loops.
    const char* WpB   = (const char*)(Wpack + (size_t)dir * (KK * G4));
    const int   boff0 = wave * 16384 + lane * 32;              // B frag base (bytes)

    const _Float16* abase = &Asm[c0 * ASTRIDE + kb_a];         // A frag base
    _Float16*       hbase = &Asm[rowb * ASTRIDE + 256 + wave * 16 + c0];

    const int ob0 = rowb * (SS * 512 * 2) + (wave * 16 + c0) * 2;  // outr lane offs

    const float* bias = biasws + dir * G4;
    float bi[2], bf[2], bg[2], bo[2];
    #pragma unroll
    for (int mm = 0; mm < 2; ++mm) {
        const int j = wave * 16 + c0 + 128 * mm;
        bi[mm] = bias[j];
        bf[mm] = bias[j + 256];
        bg[mm] = bias[j + 512];
        bo[mm] = bias[j + 768];
    }

    // cell state in registers
    float cst[RT][2][8];
    #pragma unroll
    for (int rt = 0; rt < RT; ++rt)
        #pragma unroll
        for (int mm = 0; mm < 2; ++mm)
            #pragma unroll
            for (int r = 0; r < 8; ++r) cst[rt][mm][r] = 0.0f;

    // zero the h half of the A panel (h0 = 0)
    #pragma unroll
    for (int it = 0; it < 8; ++it) {
        const int idx = (it * 256 + t) * 4;            // 0..8191
        const int row = idx >> 8;
        const int col = idx & 255;
        h4 z; z[0] = (_Float16)0.f; z[1] = (_Float16)0.f;
              z[2] = (_Float16)0.f; z[3] = (_Float16)0.f;
        *(h4*)&Asm[row * ASTRIDE + 256 + col] = z;
    }

    for (int step = 0; step < SS; ++step) {
        const int s = dir ? (SS - 1 - step) : step;

        // stage x_s tile into A panel cols [0,256) as f16
        #pragma unroll
        for (int it = 0; it < 8; ++it) {
            const int idx = (it * 256 + t) * 4;
            const int row = idx >> 8;
            const int col = idx & 255;
            const float4 v = *(const float4*)(nx +
                ((size_t)(n0 + row) * SS + s) * FF + col);
            h4 hv;
            hv[0] = (_Float16)v.x; hv[1] = (_Float16)v.y;
            hv[2] = (_Float16)v.z; hv[3] = (_Float16)v.w;
            *(h4*)&Asm[row * ASTRIDE + col] = hv;
        }

        // prefetch next step's x tile (one 128B chunk per thread)
        if (step + 1 < SS) {
            const int sn  = dir ? (s - 1) : (s + 1);
            const int row = t >> 3;
            const int col = (t & 7) * 32;
            __builtin_prefetch(nx + ((size_t)(n0 + row) * SS + sn) * FF + col, 0, 1);
        }
        __syncthreads();

        #pragma unroll
        for (int rt = 0; rt < RT; ++rt) {
            v8f acc[8];
            #pragma unroll
            for (int m = 0; m < 8; ++m)
                #pragma unroll
                for (int e = 0; e < 8; ++e) acc[m][e] = 0.0f;

            #pragma unroll 4
            for (int kt = 0; kt < KT; ++kt) {
                // A fragment: base + immediate offsets (two 16B DS loads)
                AFrag a;
                const _Float16* ap = abase + rt * (16 * ASTRIDE) + kt * 32;
                a.h[0] = *(const h8*)(ap);
                a.h[1] = *(const h8*)(ap + 16);

                #pragma unroll
                for (int m = 0; m < 8; ++m) {
                    // byte offset = boff0 + m*131072 + kt*1024 (imm-foldable)
                    const v16h b = *(const v16h*)(WpB +
                        (boff0 + m * 131072 + kt * 1024));
                    acc[m] = __builtin_amdgcn_wmma_f32_16x16x32_f16(
                        false, a.v, false, b, (short)0, acc[m], false, false);
                }
            }

            // LSTM cell update, entirely in registers
            char* obase = (char*)outr +
                (((size_t)n0 * SS + s) * 512 + (size_t)dir * 256) * 2;
            #pragma unroll
            for (int mm = 0; mm < 2; ++mm) {
                #pragma unroll
                for (int r = 0; r < 8; ++r) {
                    const float iv = sigf(acc[mm    ][r] + bi[mm]);
                    const float fv = sigf(acc[mm + 2][r] + bf[mm]);
                    const float gv = tanhf(acc[mm + 4][r] + bg[mm]);
                    const float ov = sigf(acc[mm + 6][r] + bo[mm]);
                    const float cn = fv * cst[rt][mm][r] + iv * gv;
                    cst[rt][mm][r] = cn;
                    const float hv = ov * tanhf(cn);
                    const _Float16 hh = (_Float16)hv;
                    // LDS h write: lane base + immediate offsets
                    hbase[rt * (16 * ASTRIDE) + r * ASTRIDE + 128 * mm] = hh;
                    // global h write: uniform base + lane offset + immediates
                    *(_Float16*)(obase + (ob0 +
                        rt * (16 * SS * 512 * 2) + r * (SS * 512 * 2) +
                        mm * 256)) = hh;
                }
            }
        }
        __syncthreads();
    }
}

// ---------------------------------------------------------------------------
// Kernel 2: per-node attention.  block = 256 threads = 8 waves; head = wave,
// s = lane -> per-head softmax over S=32 is a wave32 shuffle reduction.
// ---------------------------------------------------------------------------
__global__ __launch_bounds__(256) void k_attn(
    const float* __restrict__ nx,          // [N,S,F]
    const float* __restrict__ x,           // [N,F]
    const float* __restrict__ Wm,          // [HEADS, F+2H] = [8,768]
    const _Float16* __restrict__ outr,     // [N,S,512]
    float* __restrict__ out)               // [N,F]
{
    const int n = blockIdx.x;
    const int t = threadIdx.x;

    __shared__ float xs[FF];
    __shared__ float Ws[HEADS * 768];
    __shared__ float pm[HEADS][SS];
    __shared__ float ps[SS];

    xs[t] = x[(size_t)n * FF + t];
    #pragma unroll
    for (int it = 0; it < 24; ++it) Ws[it * 256 + t] = Wm[it * 256 + t];
    __syncthreads();

    const int h = t >> 5;     // head
    const int si = t & 31;    // source position s

    // logit = W[h,:] . [h_fwd | h_bwd | x]
    const float*    wrow = &Ws[h * 768];
    const _Float16* ih   = &outr[((size_t)n * SS + si) * 512];
    float acc = 0.0f;
    for (int d8 = 0; d8 < 64; ++d8) {
        const h8 hv = *(const h8*)(ih + d8 * 8);
        #pragma unroll
        for (int j = 0; j < 8; ++j) acc += wrow[d8 * 8 + j] * (float)hv[j];
    }
    for (int d = 0; d < 256; ++d) acc += wrow[512 + d] * xs[d];

    // leaky_relu(0.1) then softmax over s (within the wave)
    acc = acc > 0.0f ? acc : 0.1f * acc;
    float mx = acc;
    #pragma unroll
    for (int off = 16; off > 0; off >>= 1)
        mx = fmaxf(mx, __shfl_xor(mx, off, 32));
    const float e = __expf(acc - mx);
    float sum = e;
    #pragma unroll
    for (int off = 16; off > 0; off >>= 1)
        sum += __shfl_xor(sum, off, 32);

    pm[h][si] = (e / sum) * (1.0f / (float)HEADS);
    __syncthreads();

    if (t < SS) {
        float q = 0.0f;
        #pragma unroll
        for (int hh = 0; hh < HEADS; ++hh) q += pm[hh][t];
        ps[t] = q;
    }
    __syncthreads();

    // out[n, f] = sum_s ps[s] * nx[n, s, f]   (thread t = feature f)
    float accf = 0.0f;
    const float* nxp = nx + (size_t)n * SS * FF + t;
    for (int s2 = 0; s2 < SS; ++s2) accf += ps[s2] * nxp[s2 * FF];
    out[(size_t)n * FF + t] = accf;
}

// ---------------------------------------------------------------------------
// Launch
// ---------------------------------------------------------------------------
extern "C" void kernel_launch(void* const* d_in, const int* in_sizes, int n_in,
                              void* d_out, int out_size, void* d_ws, size_t ws_size,
                              hipStream_t stream) {
    (void)in_sizes; (void)n_in; (void)out_size; (void)ws_size;

    const float* nx    = (const float*)d_in[0];
    const float* x     = (const float*)d_in[1];
    const float* Wih_f = (const float*)d_in[2];
    const float* Whh_f = (const float*)d_in[3];
    const float* bih_f = (const float*)d_in[4];
    const float* bhh_f = (const float*)d_in[5];
    const float* Wih_r = (const float*)d_in[6];
    const float* Whh_r = (const float*)d_in[7];
    const float* bih_r = (const float*)d_in[8];
    const float* bhh_r = (const float*)d_in[9];
    const float* Wm    = (const float*)d_in[10];
    float* out = (float*)d_out;

    // workspace layout
    char* ws = (char*)d_ws;
    _Float16* Wp     = (_Float16*)ws;                       // 2 * 512*1024 halves = 2 MB
    float*    biasws = (float*)(ws + 2u * KK * G4 * 2u);    // 2 * 1024 f32
    _Float16* outr   = (_Float16*)(ws + 2105344u);          // N*S*512 halves = 256 MB

    k_prep<<<dim3((KK * G4) / 256, 2), 256, 0, stream>>>(
        Wih_f, Whh_f, bih_f, bhh_f, Wih_r, Whh_r, bih_r, bhh_r, Wp, biasws);

    k_lstm<<<dim3(NN / TM, 2), 256, 0, stream>>>(nx, Wp, biasws, outr);

    k_attn<<<NN, 256, 0, stream>>>(nx, x, Wm, outr, out);
}
